// HeteroGNN_24575802867741
// MI455X (gfx1250) — matrix operand order repaired and checked by reading, not compiled, not run on hardware
//
#include <hip/hip_runtime.h>

// ---------------------------------------------------------------------------
// Types for CDNA5 WMMA (wave32, v_wmma_f32_16x16x32_bf16)
// ---------------------------------------------------------------------------
typedef __bf16 bf16;
typedef __attribute__((ext_vector_type(16))) __bf16 v16bf;
typedef __attribute__((ext_vector_type(8)))  __bf16 v8bf;
typedef __attribute__((ext_vector_type(4)))  __bf16 v4bf;
typedef __attribute__((ext_vector_type(8)))  float  v8f;

// Problem constants (match reference)
constexpr int N_  = 25000;
constexpr int E_  = 200000;
constexpr int H_  = 512;
constexpr int OUT_= 250;
constexpr int L_  = 4;
constexpr int R_  = 3;
constexpr int HH_ = H_ * H_;

// ---------------------------------------------------------------------------
// Elementwise helpers
// ---------------------------------------------------------------------------
__global__ void k_f32_to_bf16(const float* __restrict__ in, bf16* __restrict__ out, size_t n4) {
    for (size_t i = (size_t)blockIdx.x * blockDim.x + threadIdx.x; i < n4;
         i += (size_t)gridDim.x * blockDim.x) {
        float4 v = ((const float4*)in)[i];
        v4bf o; o[0] = (bf16)v.x; o[1] = (bf16)v.y; o[2] = (bf16)v.z; o[3] = (bf16)v.w;
        ((v4bf*)out)[i] = o;
    }
}

__global__ void k_zero_f32(float* __restrict__ p, size_t n4) {
    float4 z = {0.f, 0.f, 0.f, 0.f};
    for (size_t i = (size_t)blockIdx.x * blockDim.x + threadIdx.x; i < n4;
         i += (size_t)gridDim.x * blockDim.x)
        ((float4*)p)[i] = z;
}

// Transposed bf16 weight: WT[n*K + k] = W[k*Nsrc + n], zero-padded rows n>=Nsrc.
__global__ void k_wt(const float* __restrict__ W, bf16* __restrict__ WT,
                     int K, int Nsrc, int Nn) {
    size_t total = (size_t)Nn * K;
    for (size_t i = (size_t)blockIdx.x * blockDim.x + threadIdx.x; i < total;
         i += (size_t)gridDim.x * blockDim.x) {
        int n = (int)(i / K);
        int k = (int)(i % K);
        float v = (n < Nsrc) ? W[(size_t)k * Nsrc + n] : 0.0f;
        WT[i] = (bf16)v;
    }
}

__global__ void k_deg(const int* __restrict__ dst, float* __restrict__ deg, int E) {
    for (int e = blockIdx.x * blockDim.x + threadIdx.x; e < E; e += gridDim.x * blockDim.x)
        atomicAdd(&deg[dst[e]], 1.0f);
}

__global__ void k_recip(const float* __restrict__ deg, float* __restrict__ inv, int n) {
    for (int i = blockIdx.x * blockDim.x + threadIdx.x; i < n; i += gridDim.x * blockDim.x)
        inv[i] = 1.0f / fmaxf(deg[i], 1.0f);
}

// Per-edge scatter-add of bf16 rows into f32 accumulator. One block per edge.
__global__ __launch_bounds__(128)
void k_scatter(const bf16* __restrict__ h, const int* __restrict__ src,
               const int* __restrict__ dst, float* __restrict__ agg, int H) {
    const int e = blockIdx.x;
    const int s = src[e];
    const int d = dst[e];
    const bf16* hp = h + (size_t)s * H;
    float* ap = agg + (size_t)d * H;
    for (int i = threadIdx.x * 4; i < H; i += 128 * 4) {
        v4bf hv = *(const v4bf*)(hp + i);
        atomicAdd(ap + i + 0, (float)hv[0]);
        atomicAdd(ap + i + 1, (float)hv[1]);
        atomicAdd(ap + i + 2, (float)hv[2]);
        atomicAdd(ap + i + 3, (float)hv[3]);
    }
}

// out_bf16[n*H+i] = agg[n*H+i] * inv[n].  One block per node.
__global__ __launch_bounds__(128)
void k_scale_bf16(const float* __restrict__ agg, const float* __restrict__ inv,
                  bf16* __restrict__ out, int H) {
    const int n = blockIdx.x;
    const float s = inv[n];
    const float4* ap = (const float4*)(agg + (size_t)n * H);
    v4bf* op = (v4bf*)(out + (size_t)n * H);
    for (int i = threadIdx.x; i < H / 4; i += 128) {
        float4 v = ap[i];
        v4bf o;
        o[0] = (bf16)(v.x * s); o[1] = (bf16)(v.y * s);
        o[2] = (bf16)(v.z * s); o[3] = (bf16)(v.w * s);
        op[i] = o;
    }
}

// out_bf16 = leaky(acc * scale), vectorized x4
__global__ void k_leaky(const float* __restrict__ acc, bf16* __restrict__ out,
                        float scale, size_t n4) {
    for (size_t i = (size_t)blockIdx.x * blockDim.x + threadIdx.x; i < n4;
         i += (size_t)gridDim.x * blockDim.x) {
        float4 v = ((const float4*)acc)[i];
        float a = v.x * scale, b = v.y * scale, c = v.z * scale, d = v.w * scale;
        a = (a >= 0.f) ? a : 0.2f * a;
        b = (b >= 0.f) ? b : 0.2f * b;
        c = (c >= 0.f) ? c : 0.2f * c;
        d = (d >= 0.f) ? d : 0.2f * d;
        v4bf o; o[0] = (bf16)a; o[1] = (bf16)b; o[2] = (bf16)c; o[3] = (bf16)d;
        ((v4bf*)out)[i] = o;
    }
}

// ---------------------------------------------------------------------------
// WMMA bf16 GEMM:   C[M x Nn] = A[M x K] * WT^T   (WT stored Nn x K)
// Block: 256 threads = 8 waves. Block tile 128(M) x 128(N), K-step 32.
// Waves arranged 4(M) x 2(N); each wave owns 32x64 -> 2 A-frags x 4 B-frags
// = 8 v_wmma_f32_16x16x32_bf16 per K-step.  Double-buffered LDS, one
// barrier per K-step (WMMAs consume the previous buffer before the wave
// signals the barrier, so post-barrier stores to the other buffer are safe).
// ---------------------------------------------------------------------------
__global__ __launch_bounds__(256)
void k_gemm(const bf16* __restrict__ A, const bf16* __restrict__ WT,
            const float* __restrict__ bias, float* __restrict__ accOut,
            bf16* __restrict__ bOut, int M, int K, int Nn, int NnStore,
            int accumulate, float scale)
{
    __shared__ bf16 sA[2][128 * 32];   // A tile, row-major (M x K)
    __shared__ bf16 sB[2][128 * 32];   // WT tile, row-major (N x K)

    const int tid  = threadIdx.x;
    const int wave = tid >> 5;
    const int lane = tid & 31;
    const int lr   = lane & 15;     // row-within-16 (M for A frag, N for B frag)
    const int lh   = lane >> 4;     // half-wave select
    const int wm   = (wave & 3) * 32;   // wave M offset in block tile
    const int wn   = (wave >> 2) * 64;  // wave N offset in block tile
    const int m0 = blockIdx.x * 128;
    const int n0 = blockIdx.y * 128;

    // Cooperative staging: 256 rows total (128 A + 128 B), 64B (one K-chunk) per thread.
    const int  srow = tid & 127;
    const bool isB  = tid >= 128;
    const bf16* gptr;
    if (!isB) { int r = m0 + srow; if (r >= M)  r = M - 1;  gptr = A  + (size_t)r * K; }
    else      { int r = n0 + srow; if (r >= Nn) r = Nn - 1; gptr = WT + (size_t)r * K; }
    const int soff = srow * 32;

    v8f zero = {0.f,0.f,0.f,0.f,0.f,0.f,0.f,0.f};
    v8f accv[2][4] = {{zero, zero, zero, zero}, {zero, zero, zero, zero}};

    auto stage = [&](int b, uint4 q0, uint4 q1, uint4 q2, uint4 q3) {
        bf16* s = (isB ? sB[b] : sA[b]) + soff;
        *(uint4*)(s)      = q0;
        *(uint4*)(s + 8)  = q1;
        *(uint4*)(s + 16) = q2;
        *(uint4*)(s + 24) = q3;
    };

    auto compute = [&](int b) {
        // A fragments (16x32 bf16): lane lr = M row; K = lh*16 .. lh*16+15
        v16bf af[2];
        #pragma unroll
        for (int mt = 0; mt < 2; ++mt) {
            const bf16* pa = sA[b] + (wm + mt * 16 + lr) * 32;
            v8bf alo = *(const v8bf*)(pa + lh * 16);
            v8bf ahi = *(const v8bf*)(pa + lh * 16 + 8);
            af[mt] = __builtin_shufflevector(alo, ahi,
                         0,1,2,3,4,5,6,7,8,9,10,11,12,13,14,15);
        }
        #pragma unroll
        for (int nt = 0; nt < 4; ++nt) {
            // B fragment (32x16 bf16): lane lr = N col; K = lh*16 .. lh*16+15
            v16bf bfg = *(const v16bf*)(sB[b] + (wn + nt * 16 + lr) * 32 + lh * 16);
            #pragma unroll
            for (int mt = 0; mt < 2; ++mt) {
                accv[mt][nt] = __builtin_amdgcn_wmma_f32_16x16x32_bf16(
                    false, af[mt], false, bfg, (short)0, accv[mt][nt], false, false);
            }
        }
    };

    // Prologue: stage tile 0
    {
        uint4 q0 = *(const uint4*)(gptr + 0);
        uint4 q1 = *(const uint4*)(gptr + 8);
        uint4 q2 = *(const uint4*)(gptr + 16);
        uint4 q3 = *(const uint4*)(gptr + 24);
        stage(0, q0, q1, q2, q3);
    }

    int buf = 0;
    for (int k0 = 32; k0 < K; k0 += 32) {
        uint4 q0 = *(const uint4*)(gptr + k0);
        uint4 q1 = *(const uint4*)(gptr + k0 + 8);
        uint4 q2 = *(const uint4*)(gptr + k0 + 16);
        uint4 q3 = *(const uint4*)(gptr + k0 + 24);
        __syncthreads();                 // buf ready; other buf fully consumed
        stage(buf ^ 1, q0, q1, q2, q3);  // overlaps with WMMAs below
        compute(buf);
        buf ^= 1;
    }
    __syncthreads();
    compute(buf);

    // ---------------- Epilogue ----------------
    // C/D layout: VGPR g, lanes 0-15: M=g,N=lane; 16-31: M=8+g,N=lane-16
    const bool fullM = (m0 + 128 <= M);
    const bool fullN = (n0 + 128 <= NnStore);

    if (fullM && fullN) {
        // Fast path: no bounds checks; stores are 16-lane contiguous 64B runs.
        #pragma unroll
        for (int mt = 0; mt < 2; ++mt) {
            const int mbase = m0 + wm + mt * 16 + lh * 8;
            #pragma unroll
            for (int nt = 0; nt < 4; ++nt) {
                const int n = n0 + wn + nt * 16 + lr;
                const float bv = bias ? bias[n] : 0.0f;
                const size_t base = (size_t)mbase * NnStore + n;
                #pragma unroll
                for (int g = 0; g < 8; ++g) {
                    float v = accv[mt][nt][g] + bv;
                    const size_t idx = base + (size_t)g * NnStore;
                    if (accOut) {
                        if (accumulate) v += accOut[idx];
                        accOut[idx] = v;
                    }
                    if (bOut) {
                        float o = v * scale;
                        o = (o >= 0.0f) ? o : 0.2f * o;
                        bOut[idx] = (bf16)o;
                    }
                }
            }
        }
    } else {
        #pragma unroll
        for (int mt = 0; mt < 2; ++mt) {
            #pragma unroll
            for (int nt = 0; nt < 4; ++nt) {
                int n = n0 + wn + nt * 16 + lr;
                if (n >= NnStore) continue;
                const float bv = bias ? bias[n] : 0.0f;
                #pragma unroll
                for (int g = 0; g < 8; ++g) {
                    int m = m0 + wm + mt * 16 + lh * 8 + g;
                    if (m >= M) continue;
                    float v = accv[mt][nt][g] + bv;
                    size_t idx = (size_t)m * NnStore + n;
                    if (accOut) {
                        if (accumulate) v += accOut[idx];
                        accOut[idx] = v;
                    }
                    if (bOut) {
                        float o = v * scale;
                        o = (o >= 0.0f) ? o : 0.2f * o;
                        bOut[idx] = (bf16)o;
                    }
                }
            }
        }
    }
}

// ---------------------------------------------------------------------------
// Host orchestration
// ---------------------------------------------------------------------------
static inline void launch_gemm(const bf16* A, const bf16* WT, const float* bias,
                               float* accOut, bf16* bOut, int M, int K, int Nn,
                               int NnStore, int accumulate, float scale,
                               hipStream_t stream) {
    dim3 grid((M + 127) / 128, Nn / 128);
    k_gemm<<<grid, 256, 0, stream>>>(A, WT, bias, accOut, bOut,
                                     M, K, Nn, NnStore, accumulate, scale);
}

extern "C" void kernel_launch(void* const* d_in, const int* in_sizes, int n_in,
                              void* d_out, int out_size, void* d_ws, size_t ws_size,
                              hipStream_t stream) {
    const float* x     = (const float*)d_in[0];
    const float* W_pre = (const float*)d_in[1];
    const float* W_post= (const float*)d_in[2];
    const float* Wl    = (const float*)d_in[3];
    const float* Wr    = (const float*)d_in[4];
    const float* bl    = (const float*)d_in[5];
    const float* W_out = (const float*)d_in[6];
    const float* b_out = (const float*)d_in[7];
    const int*   edges = (const int*)d_in[8];
    float* out = (float*)d_out;

    const size_t NH = (size_t)N_ * H_;

    // Workspace carve-out (256B aligned slices)
    char* wp = (char*)d_ws;
    auto carve = [&](size_t bytes) {
        char* p = wp;
        wp += (bytes + 255) & ~(size_t)255;
        return p;
    };
    bf16*  hb     = (bf16*)carve(NH * 2);
    bf16*  hb2    = (bf16*)carve(NH * 2);
    bf16*  aggb   = (bf16*)carve(NH * 2);
    float* acc    = (float*)carve(NH * 4);
    float* agg    = (float*)carve(NH * 4);
    float* invdeg = (float*)carve((size_t)R_ * N_ * 4);
    float* degtmp = (float*)carve((size_t)R_ * N_ * 4);
    bf16*  WTpre  = (bf16*)carve((size_t)HH_ * 2);
    bf16*  WTpost = (bf16*)carve((size_t)HH_ * 2);
    bf16*  WTl    = (bf16*)carve((size_t)L_ * R_ * HH_ * 2);
    bf16*  WTr    = (bf16*)carve((size_t)L_ * R_ * HH_ * 2);
    bf16*  WTout  = (bf16*)carve((size_t)256 * H_ * 2);

    // 1) Convert node features + all weights (transposed) to bf16 once.
    k_f32_to_bf16<<<4096, 256, 0, stream>>>(x, hb, NH / 4);
    k_wt<<<512, 256, 0, stream>>>(W_pre,  WTpre,  H_, H_, H_);
    k_wt<<<512, 256, 0, stream>>>(W_post, WTpost, H_, H_, H_);
    for (int i = 0; i < L_ * R_; ++i) {
        k_wt<<<512, 256, 0, stream>>>(Wl + (size_t)i * HH_, WTl + (size_t)i * HH_, H_, H_, H_);
        k_wt<<<512, 256, 0, stream>>>(Wr + (size_t)i * HH_, WTr + (size_t)i * HH_, H_, H_, H_);
    }
    k_wt<<<512, 256, 0, stream>>>(W_out, WTout, H_, OUT_, 256);

    // 2) Per-relation inverse in-degree (constant across layers).
    k_zero_f32<<<256, 256, 0, stream>>>(degtmp, (size_t)R_ * N_ / 4);
    for (int r = 0; r < R_; ++r) {
        const int* dstp = edges + (size_t)r * 2 * E_ + E_;
        k_deg<<<(E_ + 255) / 256, 256, 0, stream>>>(dstp, degtmp + (size_t)r * N_, E_);
    }
    k_recip<<<(R_ * N_ + 255) / 256, 256, 0, stream>>>(degtmp, invdeg, R_ * N_);

    // 3) Pre/post transform: h = leaky(leaky(x@Wpre)@Wpost), kept in bf16.
    launch_gemm(hb,  WTpre,  nullptr, nullptr, hb2, N_, H_, H_, H_, 0, 1.0f, stream);
    launch_gemm(hb2, WTpost, nullptr, nullptr, hb,  N_, H_, H_, H_, 0, 1.0f, stream);

    // 4) Layers.
    for (int l = 0; l < L_; ++l) {
        int first = 1;
        for (int r = 0; r < R_; ++r) {
            const int* srcp = edges + (size_t)r * 2 * E_;
            const int* dstp = srcp + E_;
            k_zero_f32<<<8192, 256, 0, stream>>>(agg, NH / 4);
            k_scatter<<<E_, 128, 0, stream>>>(hb, srcp, dstp, agg, H_);
            k_scale_bf16<<<N_, 128, 0, stream>>>(agg, invdeg + (size_t)r * N_, aggb, H_);

            const int wi = l * R_ + r;
            // acc (+)= agg@Wl + bl
            launch_gemm(aggb, WTl + (size_t)wi * HH_, bl + (size_t)wi * H_,
                        acc, nullptr, N_, H_, H_, H_, first ? 0 : 1, 1.0f, stream);
            first = 0;
            // acc += h@Wr
            launch_gemm(hb, WTr + (size_t)wi * HH_, nullptr,
                        acc, nullptr, N_, H_, H_, H_, 1, 1.0f, stream);
        }
        // h = leaky(acc / R)
        k_leaky<<<8192, 256, 0, stream>>>(acc, hb, 1.0f / (float)R_, NH / 4);
    }

    // 5) Output projection: out = h @ W_out + b_out  (f32, 25000 x 250)
    launch_gemm(hb, WTout, b_out, out, nullptr, N_, H_, 256, OUT_, 0, 1.0f, stream);
}
